// newHGalerkinConv_80350248174046
// MI455X (gfx1250) — matrix operand order repaired and checked by reading, not compiled, not run on hardware
//
#include <hip/hip_runtime.h>
#include <hip/hip_bf16.h>

// ---------------------------------------------------------------------------
// Types for WMMA fragments (wave32, CDNA5)
// ---------------------------------------------------------------------------
typedef __attribute__((ext_vector_type(2)))  __bf16 bf16x2;
typedef __attribute__((ext_vector_type(8)))  __bf16 bf16x8;
typedef __attribute__((ext_vector_type(16))) __bf16 bf16x16;
typedef __attribute__((ext_vector_type(4)))  float  f32x4;
typedef __attribute__((ext_vector_type(8)))  float  f32x8;

__device__ __forceinline__ bf16x16 cat16(bf16x8 a, bf16x8 b) {
    return __builtin_shufflevector(a, b, 0,1,2,3,4,5,6,7,8,9,10,11,12,13,14,15);
}

#define KT     64
#define LDSS   72                     // LDS row stride in bf16 (64 + 8 pad)

// ---------------------------------------------------------------------------
// Stage one 128 x 64 operand tile (f32 -> bf16) into LDS.
//   element : G[g_base + r*r_stride + k*k_stride], rows r = rbase..rbase+127,
//             k = k0..k0+63.  Layout in LDS: S[r][k], row stride LDSS.
//   k_stride == 1 : rows are 16B-aligned (all callers guarantee r_stride and
//                   k0 multiples of 4) -> 2x global b128 + 1x ds_store_b128
//                   per 8 elements, coalesced over k.
//   else          : r-fastest scalar mapping (coalesced over r, stride 1).
// ---------------------------------------------------------------------------
__device__ __forceinline__
void stage_tile(__bf16* __restrict__ S, const float* __restrict__ G,
                long g_base, long rbase, long r_stride, long k_stride,
                int k0, int tid)
{
    if (k_stride == 1) {
        const float* gp = G + g_base + k0;
        #pragma unroll
        for (int it = 0; it < 4; ++it) {
            int q  = tid + it * 256;        // 1024 groups of 8
            int r  = q >> 3;                // 8 groups per row
            int k8 = (q & 7) * 8;
            const f32x4* src = (const f32x4*)(gp + (rbase + r) * r_stride + k8);
            f32x4 v0 = src[0];
            f32x4 v1 = src[1];
            bf16x8 pk = { (__bf16)v0.x, (__bf16)v0.y, (__bf16)v0.z, (__bf16)v0.w,
                          (__bf16)v1.x, (__bf16)v1.y, (__bf16)v1.z, (__bf16)v1.w };
            *(bf16x8*)&S[r * LDSS + k8] = pk;
        }
    } else {
        #pragma unroll
        for (int it = 0; it < 32; ++it) {
            int idx = tid + it * 256;       // 8192 elements
            int r  = idx & 127;
            int kk = idx >> 7;
            float v = G[g_base + (rbase + r) * r_stride + (long)(k0 + kk) * k_stride];
            S[r * LDSS + kk] = (__bf16)v;
        }
    }
}

// ---------------------------------------------------------------------------
// One K-step of WMMA work from staged LDS tiles (16 WMMAs per wave).
// ---------------------------------------------------------------------------
__device__ __forceinline__
void compute_step(const __bf16* __restrict__ Ac, const __bf16* __restrict__ Bc,
                  f32x8 (&acc)[2][4], int wr, int wc, int half, int l15)
{
    #pragma unroll
    for (int ks = 0; ks < KT; ks += 32) {
        // A fragments: e0..7 -> k = ks + half*8 + e ; e8..15 -> +16
        bf16x16 afr[2];
        #pragma unroll
        for (int tr = 0; tr < 2; ++tr) {
            const __bf16* rp = &Ac[(wr * 32 + tr * 16 + l15) * LDSS + ks];
            bf16x8 c0 = *(const bf16x8*)(rp + half * 8);
            bf16x8 c1 = *(const bf16x8*)(rp + 16 + half * 8);
            afr[tr] = cat16(c0, c1);
        }
        // B fragments: column n = l15, e0..7 -> k = ks + half*8 + e ; e8..15 -> +16
        bf16x16 bfr[4];
        #pragma unroll
        for (int tc = 0; tc < 4; ++tc) {
            const __bf16* rp = &Bc[(wc * 64 + tc * 16 + l15) * LDSS + ks];
            bf16x8 c0 = *(const bf16x8*)(rp + half * 8);
            bf16x8 c1 = *(const bf16x8*)(rp + 16 + half * 8);
            bfr[tc] = cat16(c0, c1);
        }
        #pragma unroll
        for (int tr = 0; tr < 2; ++tr)
            #pragma unroll
            for (int tc = 0; tc < 4; ++tc)
                acc[tr][tc] = __builtin_amdgcn_wmma_f32_16x16x32_bf16(
                    false, afr[tr], false, bfr[tc],
                    (short)0, acc[tr][tc], false, false);
    }
}

// ---------------------------------------------------------------------------
// Generic batched strided split-K GEMM:  C[m,n] = sum_k A[m,k] * B[k,n]
//   A element : A[a_base + m*a_rs + k*a_cs]
//   B element : B[b_base + k*b_rs + n*b_cs]
//   C element : C[c_base + m*c_rs + n]            (C col stride == 1 always)
//   blockIdx.z = (zb*Jdiv + zj)*kchunks + kc
//   *_base = zb * *_zb + zj * *_zj ;  c_base += kc * p_stride
//   K range for this block: [kc*Klen, (kc+1)*Klen), Klen multiple of 64.
// Block = 128x128 C tile, 256 threads (8 waves), K-step = 64, bf16 WMMA,
// double-buffered LDS (next-tile global loads overlap current WMMAs),
// last iteration peeled so the steady-state loop is branch-free.
// ---------------------------------------------------------------------------
__global__ __launch_bounds__(256)
void wmma_gemm_bf16(const float* __restrict__ Ag,
                    const float* __restrict__ Bg,
                    float* __restrict__ Cg,
                    long a_rs, long a_cs,
                    long b_rs, long b_cs,
                    long c_rs,
                    int  Klen, int kchunks, long p_stride,
                    long a_zb, long a_zj,
                    long b_zb, long b_zj,
                    long c_zb, long c_zj,
                    int  Jdiv)
{
    __shared__ __bf16 As[2][128 * LDSS];   // [buf][m][k]
    __shared__ __bf16 Bs[2][128 * LDSS];   // [buf][n][k]  (B staged transposed)

    const int tid  = threadIdx.x;
    const int lane = tid & 31;
    const int wave = tid >> 5;
    const int wr   = wave >> 1;         // 0..3 : wave row  (32 rows each)
    const int wc   = wave & 1;          // 0..1 : wave col  (64 cols each)
    const int half = lane >> 4;         // 0/1  : lane half
    const int l15  = lane & 15;

    const long blockRow = (long)blockIdx.x * 128;
    const long blockCol = (long)blockIdx.y * 128;

    const int z  = blockIdx.z;
    const int kc = z % kchunks;
    const int zz = z / kchunks;
    const int zj = zz % Jdiv;
    const int zb = zz / Jdiv;
    const long a_base = (long)zb * a_zb + (long)zj * a_zj;
    const long b_base = (long)zb * b_zb + (long)zj * b_zj;
    const long c_base = (long)zb * c_zb + (long)zj * c_zj + (long)kc * p_stride;

    const int kbeg   = kc * Klen;
    const int nsteps = Klen / KT;

    f32x8 acc[2][4] = {};

    // prologue: stage first tile into buffer 0
    stage_tile(As[0], Ag, a_base, blockRow, a_rs, a_cs, kbeg, tid);
    stage_tile(Bs[0], Bg, b_base, blockCol, b_cs, b_rs, kbeg, tid);

    int p = 0;
    for (int s = 0; s < nsteps - 1; ++s, p ^= 1) {
        __syncthreads();   // buffer p fully staged; buffer p^1 free to refill
        int kn = kbeg + (s + 1) * KT;
        stage_tile(As[p ^ 1], Ag, a_base, blockRow, a_rs, a_cs, kn, tid);
        stage_tile(Bs[p ^ 1], Bg, b_base, blockCol, b_cs, b_rs, kn, tid);
        compute_step(As[p], Bs[p], acc, wr, wc, half, l15);
    }
    __syncthreads();
    compute_step(As[p], Bs[p], acc, wr, wc, half, l15);   // epilogue step

    // ---- store C: lane holds (r + 8*half, l15) of each 16x16 tile ----
    #pragma unroll
    for (int tr = 0; tr < 2; ++tr) {
        #pragma unroll
        for (int tc = 0; tc < 4; ++tc) {
            long gr = blockRow + wr * 32 + tr * 16;
            long gc = blockCol + wc * 64 + tc * 16;
            #pragma unroll
            for (int r = 0; r < 8; ++r)
                Cg[c_base + (gr + r + 8 * half) * c_rs + gc + l15] = acc[tr][tc][r];
        }
    }
}

// ---------------------------------------------------------------------------
// Deterministic split-K reduction:  out[i] = sum_c in[i + c*stride]
// ---------------------------------------------------------------------------
__global__ __launch_bounds__(256)
void reduce_sum(const float* __restrict__ in, float* __restrict__ out_,
                int chunks, long stride)
{
    long i = (long)blockIdx.x * 256 + threadIdx.x;
    float s = 0.f;
    for (int c = 0; c < chunks; ++c) s += in[i + c * stride];
    out_[i] = s;
}

// ---------------------------------------------------------------------------
// Build transposed H:  Ht[j][l][k] = D_out[j,k]*D_in[j,l]*product[k,l]
//                                    + (k<64 && l<64 ? sum_r A[j,r,k]*B[j,r,l] : 0)
// idx = j*16384 + l*128 + k  (262144 elements)
// ---------------------------------------------------------------------------
__global__ __launch_bounds__(256)
void build_Ht(const float* __restrict__ D_out, const float* __restrict__ D_in,
              const float* __restrict__ product,
              const float* __restrict__ Amat, const float* __restrict__ Bmat,
              float* __restrict__ Ht)
{
    int idx = blockIdx.x * 256 + threadIdx.x;
    int k = idx & 127;
    int l = (idx >> 7) & 127;
    int j = idx >> 14;
    float v = D_out[j * 128 + k] * D_in[j * 128 + l] * product[k * 128 + l];
    if (k < 64 && l < 64) {
        float q = 0.f;
        #pragma unroll
        for (int r = 0; r < 8; ++r)
            q += Amat[(j * 8 + r) * 64 + k] * Bmat[(j * 8 + r) * 64 + l];
        v += q;
    }
    Ht[idx] = v;
}

// ---------------------------------------------------------------------------
// Pre-transpose weights (DIM,DIM,J) -> Wt[j][o][i]  (coalesced writes)
// ---------------------------------------------------------------------------
__global__ __launch_bounds__(256)
void transpose_w(const float* __restrict__ w, float* __restrict__ wt)
{
    int idx = blockIdx.x * 256 + threadIdx.x;   // 262144: j*16384 + o*128 + i
    int i = idx & 127;
    int o = (idx >> 7) & 127;
    int j = idx >> 14;
    wt[idx] = w[i * 2048 + o * 16 + j];
}

// ---------------------------------------------------------------------------
// Orchestration.
// ws (f32 elems): x_hat @0 (262144) | Ht @262144 (262144) | Wt @524288 (262144)
//                 T @786432 (4194304, [b][o][j][l]) | y @4980736 (262144)
// d_out doubles as split-K partial scratch before its final full overwrite.
// ---------------------------------------------------------------------------
extern "C" void kernel_launch(void* const* d_in, const int* in_sizes, int n_in,
                              void* d_out, int out_size, void* d_ws, size_t ws_size,
                              hipStream_t stream)
{
    (void)in_sizes; (void)n_in; (void)out_size; (void)ws_size;

    const float* x       = (const float*)d_in[0];   // (16,128,16384)
    const float* bases   = (const float*)d_in[1];   // (16384,128)
    const float* wbases  = (const float*)d_in[2];   // (16384,128)
    const float* product = (const float*)d_in[3];   // (128,128)
    const float* D_out   = (const float*)d_in[4];   // (16,128)
    const float* D_in    = (const float*)d_in[5];   // (16,128)
    const float* Amat    = (const float*)d_in[6];   // (16,8,64)
    const float* Bmat    = (const float*)d_in[7];   // (16,8,64)
    const float* weights = (const float*)d_in[8];   // (128,128,16)
    float* out  = (float*)d_out;                    // (16,128,16384)
    float* pbuf = (float*)d_out;                    // split-K partial scratch

    float* ws    = (float*)d_ws;
    float* x_hat = ws;                 // [b][i][l]   (2048 x 128)
    float* Ht    = ws + 262144;        // [j][l][k]
    float* Wt    = ws + 524288;        // [j][o][i]
    float* T     = ws + 786432;        // [b][o][j][l]
    float* y     = ws + 4980736;       // [b][o][k]   (2048 x 128)

    dim3 blk(256);

    // 1) x_hat[(b,i), l] = sum_x x[(b,i), x] * wbases[x, l]
    //    M=2048 N=128 Kr=16384, split-K 32x512 -> 512 blocks, partials in d_out
    wmma_gemm_bf16<<<dim3(16, 1, 32), blk, 0, stream>>>(
        x, wbases, pbuf,
        16384L, 1L,          // A: x (k contiguous -> b128 packed staging)
        128L,   1L,          // B: wbases (n stride 1 -> n-fast staging)
        128L,
        /*Klen*/512, /*kchunks*/32, /*p_stride*/262144L,
        0L, 0L, 0L, 0L, 0L, 0L, 1);
    reduce_sum<<<dim3(1024), blk, 0, stream>>>(pbuf, x_hat, 32, 262144L);

    // 2) small precomputes
    build_Ht<<<dim3(1024), blk, 0, stream>>>(D_out, D_in, product, Amat, Bmat, Ht);
    transpose_w<<<dim3(1024), blk, 0, stream>>>(weights, Wt);

    // 3) T[b][o][j][l] = sum_i Wt[j,o,i] * x_hat[b,i,l]
    //    z = b*16 + j (256 blocks); M=o N=l Kr=i=128
    wmma_gemm_bf16<<<dim3(1, 1, 256), blk, 0, stream>>>(
        Wt, x_hat, T,
        128L, 1L,            // A: Wt (k=i contiguous -> packed staging)
        128L, 1L,            // B: x_hat
        2048L,
        /*Klen*/128, /*kchunks*/1, /*p_stride*/0L,
        /*a_zb*/0L, /*a_zj*/16384L,
        /*b_zb*/16384L, /*b_zj*/0L,
        /*c_zb*/262144L, /*c_zj*/128L, 16);

    // 4) y[b][o][k] = sum_p T[b][o][p] * Ht[p][k]   (p = j*128+l, 2048)
    //    split-K 8x256 -> 128 blocks, partials in d_out
    wmma_gemm_bf16<<<dim3(1, 1, 128), blk, 0, stream>>>(
        T, Ht, pbuf,
        2048L, 1L,           // A: T (k contiguous)
        128L, 1L,            // B: Ht
        128L,
        /*Klen*/256, /*kchunks*/8, /*p_stride*/262144L,
        /*a_zb*/262144L, /*a_zj*/0L,
        /*b_zb*/0L, /*b_zj*/0L,
        /*c_zb*/16384L, /*c_zj*/0L, 1);
    reduce_sum<<<dim3(1024), blk, 0, stream>>>(pbuf, y, 8, 262144L);

    // 5) out[(b,o), x] = sum_k y[(b,o), k] * bases[x, k]
    //    M=2048 N=16384 Kr=128 -> 2048 blocks (overwrites all of d_out)
    wmma_gemm_bf16<<<dim3(16, 128, 1), blk, 0, stream>>>(
        y, bases, out,
        128L, 1L,            // A: y (k contiguous)
        1L, 128L,            // B: bases (k stride 1 -> packed staging)
        16384L,
        /*Klen*/128, /*kchunks*/1, /*p_stride*/0L,
        0L, 0L, 0L, 0L, 0L, 0L, 1);
}
